// ConvSIVILayer_28183575396676
// MI455X (gfx1250) — compile-verified, matrix operand order checked by reading, not run on hardware
//
#include <hip/hip_runtime.h>
#include <hip/hip_bf16.h>
#include <cstddef>

// ---------------- problem constants ----------------
#define NMC   16
#define NB    16
#define CIN   16
#define COUT  16
#define KK    3
#define HH    128
#define WW    128
#define OH    126
#define OW    126
#define NHID  2325
#define NPAR  4640
#define WSPLIT 4608            // 16*16*9*2
#define NWK   36864            // 256*16*3*3
#define YTOT  ((size_t)NMC*NB*COUT*OH*OW)   // 65,028,096

// conv tiling
#define ROWS_PER_BLK 10
#define NTILEY       13        // ceil(126/10)
#define IN_ROWS      12        // 10 + (KK-1)
#define ICP          18        // padded ic stride (bank-conflict friendly, keeps b32 alignment)

typedef __attribute__((ext_vector_type(16))) __bf16 bf16x16;
typedef __attribute__((ext_vector_type(8)))  float  f32x8;

union BFrag { unsigned u[8]; bf16x16 v; };

static __device__ __forceinline__ unsigned short f2bf(float f) {
    unsigned u = __float_as_uint(f);
    unsigned r = (u + 0x7FFFu + ((u >> 16) & 1u)) >> 16;   // round-to-nearest-even
    return (unsigned short)r;
}

static __device__ __forceinline__ float softplusf(float x) {
    return (x > 20.f) ? x : log1pf(expf(x));
}

// ---------------- hyper-network MLP (f32 VALU, precision-critical) ----------------
__global__ void mlp_layer(const float* __restrict__ in, const float* __restrict__ Wt,
                          const float* __restrict__ bias, float* __restrict__ out,
                          int K, int N, int do_relu)
{
    int j  = blockIdx.x * blockDim.x + threadIdx.x;
    int mc = blockIdx.y;
    if (j >= N) return;
    const float* inr = in + (size_t)mc * K;
    float acc = bias[j];
    for (int k = 0; k < K; ++k)
        acc = fmaf(inr[k], Wt[(size_t)k * N + j], acc);
    if (do_relu) acc = fmaxf(acc, 0.f);
    out[(size_t)mc * N + j] = acc;
}

// ---------------- wk / bk / KL partials ----------------
// blocks 0..143: 256 wk elements each (+ KL block partial); block 144: 256 bk elements
__global__ void wkbk_kernel(const float* __restrict__ sivi, const float* __restrict__ eps_w,
                            const float* __restrict__ eps_b, float* __restrict__ wk,
                            float* __restrict__ bkv, float* __restrict__ partial)
{
    __shared__ float red[256];
    int tid = threadIdx.x;
    int blk = blockIdx.x;
    if (blk < 144) {
        int j   = blk * 256 + tid;             // 0..36863
        int mc  = j / 2304;
        int off = j - mc * 2304;
        float mu = sivi[(size_t)mc * NPAR + off];
        float sd = softplusf(sivi[(size_t)mc * NPAR + 2304 + off]);
        wk[j] = fmaf(sd, eps_w[j], mu);
        float kl = logf(sd) + (1.f + mu * mu) / (2.f * sd * sd) - 0.5f;
        red[tid] = kl;
        __syncthreads();
        for (int s = 128; s > 0; s >>= 1) {
            if (tid < s) red[tid] += red[tid + s];
            __syncthreads();
        }
        if (tid == 0) partial[blk] = red[0];
    } else {
        int mc = tid >> 4, oc = tid & 15;
        float mu = sivi[(size_t)mc * NPAR + WSPLIT + oc];
        float sd = softplusf(sivi[(size_t)mc * NPAR + WSPLIT + 16 + oc]);
        bkv[tid] = fmaf(sd, eps_b[tid], mu);
    }
}

__global__ void prior_kernel(const float* __restrict__ partial,
                             const float* __restrict__ prior_in,
                             float* __restrict__ out_prior)
{
    __shared__ float red[256];
    int tid = threadIdx.x;
    red[tid] = (tid < 144) ? partial[tid] : 0.f;
    __syncthreads();
    for (int s = 128; s > 0; s >>= 1) {
        if (tid < s) red[tid] += red[tid + s];
        __syncthreads();
    }
    if (tid == 0) out_prior[0] = prior_in[0] + red[0] * (1.0f / 256.0f);
}

// ---------------- grouped conv via WMMA bf16 ----------------
// One block = (mc, b, 10 output rows). 8 waves; wave w owns column tile w (16 cols).
// GEMM per 16x16 output tile: M=16 oc, N=16 pixels, K=144 (pad 160) in 5 wmma steps.
// K ordering: k = (kh*3+kw)*16 + ic  -> packed bf16 pair = (ic, ic+1) at one tap.
__global__ __launch_bounds__(256) void conv_kernel(const float* __restrict__ x,
                                                   const float* __restrict__ wk,
                                                   const float* __restrict__ bkv,
                                                   float* __restrict__ y)
{
    __shared__ unsigned short xs[IN_ROWS * WW * ICP];   // 55,296 B
    __shared__ unsigned short wA[10 * 16 * 16];         //  5,120 B  [pos][m][ic]
    __shared__ float          bks[16];

    const int tileY = blockIdx.x;
    const int b     = blockIdx.y;
    const int mc    = blockIdx.z;
    const int r0    = tileY * ROWS_PER_BLK;
    const int nInRows = (HH - r0 < IN_ROWS) ? (HH - r0) : IN_ROWS;
    const int tid   = threadIdx.x;

    // stage weights (f32 -> bf16), zero-pad pos==9
    for (int e = tid; e < 10 * 16 * 16; e += 256) {
        int pos = e >> 8, m = (e >> 4) & 15, ic = e & 15;
        float v = 0.f;
        if (pos < 9) v = wk[((size_t)(mc * 16 + m)) * 144 + ic * 9 + pos];
        wA[(pos * 16 + m) * 16 + ic] = f2bf(v);
    }
    if (tid < 16) bks[tid] = bkv[mc * 16 + tid];

    // stage input tile (f32 -> bf16), layout xs[r][col][ic]
    const float* xb = x + (((size_t)mc * NB + b) * CIN) * HH * WW;
    for (int e = tid; e < IN_ROWS * 16 * WW; e += 256) {
        int col = e & 127, t2 = e >> 7;
        int ic = t2 & 15, r = t2 >> 4;
        float v = 0.f;
        if (r < nInRows) v = xb[(size_t)ic * HH * WW + (size_t)(r0 + r) * WW + col];
        xs[(r * WW + col) * ICP + ic] = f2bf(v);
    }
    __syncthreads();

    const int wv   = tid >> 5;     // wave -> column tile 0..7
    const int lane = tid & 31;
    const int n    = lane & 15;    // N column (B/C/D) and M row (A)
    const int half = lane >> 4;
    const int col0 = wv * 16;

    // A fragments: 5 K-steps x 8 packed regs; k0 = 32s + (i<4 ? 2i : 16+2(i-4)) + 8*half
    unsigned afrag[5][8];
#pragma unroll
    for (int s = 0; s < 5; ++s) {
#pragma unroll
        for (int i = 0; i < 8; ++i) {
            const int pos = 2 * s + (i >= 4 ? 1 : 0);   // compile-time
            int ic = 2 * (i & 3) + 8 * half;
            afrag[s][i] = *(const unsigned*)&wA[(pos * 16 + n) * 16 + ic];
        }
    }
    float biasv[8];
#pragma unroll
    for (int r = 0; r < 8; ++r) biasv[r] = bks[r + 8 * half];

    for (int row = 0; row < ROWS_PER_BLK; ++row) {
        const int rowg = r0 + row;
        if (rowg >= OH) break;               // block-uniform
        f32x8 c;
#pragma unroll
        for (int r = 0; r < 8; ++r) c[r] = biasv[r];   // D = A*B + bias
#pragma unroll
        for (int s = 0; s < 5; ++s) {
            BFrag bf;
#pragma unroll
            for (int i = 0; i < 8; ++i) {
                const int pos = 2 * s + (i >= 4 ? 1 : 0);   // compile-time
                if (pos == 9) { bf.u[i] = 0u; continue; }   // K padding 144..159
                const int kh = pos / 3;                      // compile-time
                const int kw = pos - 3 * kh;                 // compile-time
                int ic   = 2 * (i & 3) + 8 * half;
                int colx = col0 + n + kw; if (colx > WW - 1) colx = WW - 1;
                bf.u[i] = *(const unsigned*)&xs[((row + kh) * WW + colx) * ICP + ic];
            }
            BFrag af;
#pragma unroll
            for (int i = 0; i < 8; ++i) af.u[i] = afrag[s][i];
            c = __builtin_amdgcn_wmma_f32_16x16x32_bf16(
                    false, af.v, false, bf.v, (short)0, c, false, false);
        }
        // store: lane n -> col, VGPR r -> oc = r + 8*half  (streaming -> non-temporal)
        const int colg = col0 + n;
        if (colg < OW) {
#pragma unroll
            for (int r = 0; r < 8; ++r) {
                int mm = r + 8 * half;
                __builtin_nontemporal_store(
                    c[r], &y[((((size_t)mc * NB + b) * COUT + mm) * OH + rowg) * OW + colg]);
            }
        }
    }
}

// ---------------- launch ----------------
extern "C" void kernel_launch(void* const* d_in, const int* in_sizes, int n_in,
                              void* d_out, int out_size, void* d_ws, size_t ws_size,
                              hipStream_t stream) {
    (void)in_sizes; (void)n_in; (void)out_size; (void)ws_size;
    const float* x     = (const float*)d_in[0];
    const float* prior = (const float*)d_in[1];
    const float* W1    = (const float*)d_in[2];
    const float* b1    = (const float*)d_in[3];
    const float* W2    = (const float*)d_in[4];
    const float* b2    = (const float*)d_in[5];
    const float* W3    = (const float*)d_in[6];
    const float* b3    = (const float*)d_in[7];
    const float* noise = (const float*)d_in[8];
    const float* eps_w = (const float*)d_in[9];
    const float* eps_b = (const float*)d_in[10];

    float* y = (float*)d_out;

    float* ws      = (float*)d_ws;
    float* h1      = ws;                    // 16*2325
    float* h2      = h1 + NMC * NHID;       // 16*2325
    float* sivi    = h2 + NMC * NHID;       // 16*4640
    float* wk      = sivi + NMC * NPAR;     // 36864
    float* bk      = wk + NWK;              // 256
    float* partial = bk + 256;              // 144

    mlp_layer<<<dim3((NHID + 255) / 256, NMC), 256, 0, stream>>>(noise, W1, b1, h1, 10,   NHID, 1);
    mlp_layer<<<dim3((NHID + 255) / 256, NMC), 256, 0, stream>>>(h1,    W2, b2, h2, NHID, NHID, 1);
    mlp_layer<<<dim3((NPAR + 255) / 256, NMC), 256, 0, stream>>>(h2,    W3, b3, sivi, NHID, NPAR, 0);
    wkbk_kernel<<<145, 256, 0, stream>>>(sivi, eps_w, eps_b, wk, bk, partial);
    prior_kernel<<<1, 256, 0, stream>>>(partial, prior, y + YTOT);
    conv_kernel<<<dim3(NTILEY, NB, NMC), 256, 0, stream>>>(x, wk, bk, y);
}
